// NoFieldSymmetricPredictionSourceBlock_13872744366309
// MI455X (gfx1250) — compile-verified, matrix operand order checked by reading, not run on hardware
//
#include <hip/hip_runtime.h>
#include <math.h>

typedef __attribute__((ext_vector_type(16))) __bf16 v16bf;
typedef __attribute__((ext_vector_type(8)))  float  v8f;

__device__ __forceinline__ v8f wmma_bf16(v16bf a, v16bf b, v8f c) {
  return __builtin_amdgcn_wmma_f32_16x16x32_bf16(false, a, false, b, (short)0, c,
                                                 false, false);
}

__device__ __forceinline__ v8f zero_v8f() {
  v8f z = {0.f, 0.f, 0.f, 0.f, 0.f, 0.f, 0.f, 0.f};
  return z;
}

// 16-bit A/B fragment half-index -> K offset (ISA 7.12.2, 16-bit A 16x32):
// lane<16 holds K {0..7,16..23}, lane>=16 holds K {8..15,24..31}
__device__ __forceinline__ int wmma_kidx(int h, int hi) {
  int j = h >> 1, p = h & 1;
  return ((j & 4) << 2) + (hi ? 8 : 0) + ((j & 3) << 1) + p;
}

// Load a pre-staged B fragment (16 bf16 halves per lane, contiguous) from LDS.
__device__ __forceinline__ v16bf load_bfrag(const __bf16* frag, int lane) {
  union { v16bf v; uint4 q[2]; } u;
  const uint4* p = (const uint4*)(frag + lane * 16);
  u.q[0] = p[0];
  u.q[1] = p[1];
  return u.v;
}

// Load an A fragment from a 16x64 bf16 row-major LDS activation tile.
__device__ __forceinline__ v16bf load_afrag_lds(const __bf16* act, int lane, int kc) {
  int m   = lane & 15;
  int hi  = (lane >> 4) & 1;
  int off = kc * 32 + hi * 8;                    // in halves; multiples of 8
  const uint4* row = (const uint4*)(act + m * 64);
  union { v16bf v; uint4 q[2]; } u;
  u.q[0] = row[off >> 3];
  u.q[1] = row[(off + 16) >> 3];
  return u.v;
}

// Build an A fragment from a global f32 row (contiguous channels).
__device__ __forceinline__ v16bf load_afrag_contig(const float* row, int off) {
  v16bf a;
#pragma unroll
  for (int i = 0; i < 8; ++i) a[i] = (__bf16)row[off + i];
#pragma unroll
  for (int i = 0; i < 8; ++i) a[8 + i] = (__bf16)row[off + 16 + i];
  return a;
}

__device__ __forceinline__ float silu_f(float x) { return x / (1.0f + __expf(-x)); }

// -------------------------------------------------------------------------
// Kernel 1: node-side O(3) linears with WMMA.
//   s1 = s@W1_0/8, s2 = s@W2_0/8  (s = node_feats[:, :64])
//   v1[n][c][m] = sum_k v[n,k,m] W1_1[k,c] / 8 ; v2 likewise with W2_1
// One wave per 16-node tile; 64 v_wmma per tile.
// Overlapping-tile trick: partial/duplicate tiles rewrite identical values,
// so there are no store guards at all. Requires N >= 16.
// -------------------------------------------------------------------------
__global__ void __launch_bounds__(128)
node_linear_kernel(const float* __restrict__ nf,
                   const float* __restrict__ W10, const float* __restrict__ W11,
                   const float* __restrict__ W20, const float* __restrict__ W21,
                   float* __restrict__ s1t, float* __restrict__ s2t,
                   float* __restrict__ v1t, float* __restrict__ v2t,
                   int N, int nTiles)
{
  __shared__ __align__(16) __bf16 wfrag[32][32][16];  // 32 B-fragments

  // Stage all weight fragments: f = w*8 + kc*4 + nb ; w: 0=W10,1=W11,2=W20,3=W21
  for (int idx = threadIdx.x; idx < 32 * 32; idx += blockDim.x) {
    int f = idx >> 5, lane = idx & 31;
    int w = f >> 3, kc = (f >> 2) & 1, nb = f & 3;
    const float* Wp = (w == 0) ? W10 : (w == 1) ? W11 : (w == 2) ? W20 : W21;
    int hi  = lane >> 4;
    int col = nb * 16 + (lane & 15);
#pragma unroll
    for (int h = 0; h < 16; ++h) {
      int k = kc * 32 + wmma_kidx(h, hi);
      wfrag[f][lane][h] = (__bf16)Wp[k * 64 + col];
    }
  }
  __syncthreads();

  const int lane = threadIdx.x & 31;
  const int wid  = threadIdx.x >> 5;
  const int hi   = (lane >> 4) & 1;
  int tile = blockIdx.x * 4 + wid;
  if (tile >= nTiles) tile = nTiles - 1;   // duplicate tile: identical rewrites
  int n0 = tile * 16;
  if (n0 + 16 > N) n0 = N - 16;            // overlapped tile: identical rewrites

  const float* rowbase = nf + (long)(n0 + (lane & 15)) * 256;

  // ---- scalar channels: s1, s2 ----
  {
    v16bf a0 = load_afrag_contig(rowbase, hi * 8);
    v16bf a1 = load_afrag_contig(rowbase, 32 + hi * 8);
#pragma unroll
    for (int w = 0; w < 2; ++w) {
      float* outp = w ? s2t : s1t;
      int fb = (w ? 2 : 0) * 8;
#pragma unroll
      for (int nb = 0; nb < 4; ++nb) {
        v8f acc = zero_v8f();
        acc = wmma_bf16(a0, load_bfrag(&wfrag[fb + nb][0][0], lane), acc);
        acc = wmma_bf16(a1, load_bfrag(&wfrag[fb + 4 + nb][0][0], lane), acc);
        int c = (lane & 15) + nb * 16;
        float* p = outp + (long)(n0 + 8 * hi) * 64 + c;
#pragma unroll
        for (int r = 0; r < 8; ++r) p[r * 64] = acc[r] * 0.125f;
      }
    }
  }

  // ---- vector channels: read four contiguous 24-float runs, build frags ----
  {
    const float* vb = rowbase + 64;         // v[n,k,m] = vb[3k + m]
    float run[4][24];                        // q: channel runs {0,16,32,48}+hi*8
#pragma unroll
    for (int q = 0; q < 4; ++q) {
      int cstart = ((q & 1) ? 16 : 0) + ((q >> 1) ? 32 : 0) + hi * 8;
      const float* p = vb + 3 * cstart;
#pragma unroll
      for (int i = 0; i < 24; ++i) run[q][i] = p[i];
    }
#pragma unroll
    for (int mm = 0; mm < 3; ++mm) {
      v16bf a0, a1;
#pragma unroll
      for (int i = 0; i < 8; ++i) {
        a0[i]     = (__bf16)run[0][3 * i + mm];
        a0[8 + i] = (__bf16)run[1][3 * i + mm];
        a1[i]     = (__bf16)run[2][3 * i + mm];
        a1[8 + i] = (__bf16)run[3][3 * i + mm];
      }
#pragma unroll
      for (int w = 0; w < 2; ++w) {
        float* outp = w ? v2t : v1t;
        int fb = (w ? 3 : 1) * 8;
#pragma unroll
        for (int nb = 0; nb < 4; ++nb) {
          v8f acc = zero_v8f();
          acc = wmma_bf16(a0, load_bfrag(&wfrag[fb + nb][0][0], lane), acc);
          acc = wmma_bf16(a1, load_bfrag(&wfrag[fb + 4 + nb][0][0], lane), acc);
          int c = (lane & 15) + nb * 16;
          float* p = outp + ((long)(n0 + 8 * hi) * 64 + c) * 3 + mm;
#pragma unroll
          for (int r = 0; r < 8; ++r) p[r * 192] = acc[r] * 0.125f;
        }
      }
    }
  }
}

// -------------------------------------------------------------------------
// Kernel 2: mvec (tiny dot) + zero-init of the charge slot of multipoles.
// -------------------------------------------------------------------------
__global__ void mvec_init_kernel(const float* __restrict__ nf,
                                 const float* __restrict__ Wm1,
                                 float* __restrict__ mp, int N)
{
  int n = blockIdx.x * blockDim.x + threadIdx.x;
  if (n >= N) return;
  const float* row = nf + (long)n * 256 + 64;
  float a0 = 0.f, a1 = 0.f, a2 = 0.f;
#pragma unroll 4
  for (int c = 0; c < 64; ++c) {
    float w = Wm1[c];
    a0 += row[3 * c] * w;
    a1 += row[3 * c + 1] * w;
    a2 += row[3 * c + 2] * w;
  }
  mp[(long)n * 4 + 0] = 0.f;           // charges accumulated by edge kernel
  mp[(long)n * 4 + 1] = a0 * 0.125f;
  mp[(long)n * 4 + 2] = a1 * 0.125f;
  mp[(long)n * 4 + 3] = a2 * 0.125f;
}

// -------------------------------------------------------------------------
// Kernel 3: fused fc_net (WMMA chain) + edge message + scatter.
// One wave per 16-edge tile; 36 v_wmma per tile (only w0/w3 cols of fc_w3).
// Overlapping-tile trick for the pure chain; ownership guard (e >= tileStart)
// keeps the non-idempotent atomics exactly-once. Requires E >= 16.
// -------------------------------------------------------------------------
__global__ void __launch_bounds__(128)
edge_kernel(const float* __restrict__ edge_attrs,
            const float* __restrict__ edge_feats,
            const int*   __restrict__ edge_index,
            const float* __restrict__ fw0, const float* __restrict__ fw1,
            const float* __restrict__ fw2, const float* __restrict__ fw3,
            const float* __restrict__ Wf,
            const float* __restrict__ s1t, const float* __restrict__ s2t,
            const float* __restrict__ v1t, const float* __restrict__ v2t,
            float* __restrict__ multipoles, float* __restrict__ p_out,
            int N, int E, int nTiles)
{
  // 36 B fragments: [0..3] fw0 (K padded 8->32); [4..11] fw1; [12..19] fw2;
  // [20..35] fw3 cols {0..63} (g=0) and {192..255} (g=1): f = 20 + g*8 + kc*4 + nb
  __shared__ __align__(16) __bf16 wfrag[36][32][16];
  __shared__ __align__(16) __bf16 s_act[4][16 * 64];   // per-wave activation tile
  __shared__ float wf_s[128];

  for (int idx = threadIdx.x; idx < 36 * 32; idx += blockDim.x) {
    int f = idx >> 5, lane = idx & 31;
    const float* Wp; int ncols = 64, krows = 64, kbase, nbase;
    if (f < 4)        { Wp = fw0; krows = 8; kbase = 0; nbase = f * 16; }
    else if (f < 12)  { int q = f - 4;  Wp = fw1; kbase = ((q >> 2) & 1) * 32; nbase = (q & 3) * 16; }
    else if (f < 20)  { int q = f - 12; Wp = fw2; kbase = ((q >> 2) & 1) * 32; nbase = (q & 3) * 16; }
    else              { int q = f - 20; Wp = fw3; ncols = 256; kbase = ((q >> 2) & 1) * 32;
                        nbase = (q & 3) * 16 + ((q >> 3) & 1) * 192; }
    int hi  = lane >> 4;
    int col = nbase + (lane & 15);
#pragma unroll
    for (int h = 0; h < 16; ++h) {
      int k = kbase + wmma_kidx(h, hi);
      float v = (k < krows) ? Wp[k * ncols + col] : 0.0f;
      wfrag[f][lane][h] = (__bf16)v;
    }
  }
  if (threadIdx.x < 128) wf_s[threadIdx.x] = Wf[threadIdx.x];
  __syncthreads();

  const int lane = threadIdx.x & 31;
  const int wid  = threadIdx.x >> 5;
  const int hi   = (lane >> 4) & 1;
  const int tileStart = (blockIdx.x * 4 + wid) * 16;   // may exceed E (idle wave)
  int base = tileStart;
  if (base + 16 > E) base = E - 16;                    // overlap: pure work only
  __bf16* act = &s_act[wid][0];

  // ---- layer 0: X0 (16x8, K zero-padded to 32) @ fw0, /sqrt(8), SiLU ----
  {
    v16bf a;
#pragma unroll
    for (int i = 0; i < 16; ++i) a[i] = (__bf16)0.0f;
    if (lane < 16) {                      // hi lanes hold only K>=8 -> all zero
      const float* row = edge_feats + (long)(base + lane) * 8;
#pragma unroll
      for (int i = 0; i < 8; ++i) a[i] = (__bf16)row[i];
    }
    const float s0 = 0.35355339059327373f;   // 1/sqrt(8)
#pragma unroll
    for (int nb = 0; nb < 4; ++nb) {
      v8f acc = zero_v8f();
      acc = wmma_bf16(a, load_bfrag(&wfrag[nb][0][0], lane), acc);
#pragma unroll
      for (int r = 0; r < 8; ++r)
        act[(r + 8 * hi) * 64 + (lane & 15) + nb * 16] = (__bf16)silu_f(acc[r] * s0);
    }
  }
  __syncthreads();

  // ---- layers 1, 2: X @ fw{1,2} / 8, SiLU ----
#pragma unroll
  for (int L = 0; L < 2; ++L) {
    int fb = 4 + L * 8;
    v16bf a0 = load_afrag_lds(act, lane, 0);
    v16bf a1 = load_afrag_lds(act, lane, 1);
    __syncthreads();
#pragma unroll
    for (int nb = 0; nb < 4; ++nb) {
      v8f acc = zero_v8f();
      acc = wmma_bf16(a0, load_bfrag(&wfrag[fb + nb][0][0], lane), acc);
      acc = wmma_bf16(a1, load_bfrag(&wfrag[fb + 4 + nb][0][0], lane), acc);
#pragma unroll
      for (int r = 0; r < 8; ++r)
        act[(r + 8 * hi) * 64 + (lane & 15) + nb * 16] = (__bf16)silu_f(acc[r] * 0.125f);
    }
    __syncthreads();
  }

  // ---- layer 3 (w0 = cols 0..63, w3 = cols 192..255) fused with message ----
  {
    v16bf a0 = load_afrag_lds(act, lane, 0);
    v16bf a1 = load_afrag_lds(act, lane, 1);

    // Per-edge metadata for this lane's 8 edges, hoisted out of the hot loops.
    int snd[8], rcv[8];
    float y1a[8], y1b[8], y1c[8];
#pragma unroll
    for (int r = 0; r < 8; ++r) {
      int e = base + r + 8 * hi;
      snd[r] = edge_index[e];
      rcv[r] = edge_index[E + e];
      const float* y = edge_attrs + (long)e * 4;
      y1a[r] = y[1]; y1b[r] = y[2]; y1c[r] = y[3];
    }

    float pA[8], pB[8];
#pragma unroll
    for (int r = 0; r < 8; ++r) { pA[r] = 0.f; pB[r] = 0.f; }

#pragma unroll
    for (int g = 0; g < 2; ++g) {
#pragma unroll
      for (int nb = 0; nb < 4; ++nb) {
        int fb = 20 + g * 8;
        v8f acc = zero_v8f();
        acc = wmma_bf16(a0, load_bfrag(&wfrag[fb + nb][0][0], lane), acc);
        acc = wmma_bf16(a1, load_bfrag(&wfrag[fb + 4 + nb][0][0], lane), acc);
        int c = (lane & 15) + nb * 16;
        float wfv = wf_s[g * 64 + c];
#pragma unroll
        for (int r = 0; r < 8; ++r) {
          float w = acc[r] * 0.125f;          // /sqrt(64)
          if (g == 0) {
            float bs = s1t[(long)snd[r] * 64 + c] + s2t[(long)rcv[r] * 64 + c];
            pA[r] += w * bs * wfv;
          } else {
            const float* pv1 = v1t + ((long)snd[r] * 64 + c) * 3;  // 3 contiguous
            const float* pv2 = v2t + ((long)rcv[r] * 64 + c) * 3;
            float dot = (pv1[0] + pv2[0]) * y1a[r]
                      + (pv1[1] + pv2[1]) * y1b[r]
                      + (pv1[2] + pv2[2]) * y1c[r];
            pB[r] += w * dot * wfv;
          }
        }
      }
    }

    // Reduce over the 16-lane half-groups (xor masks 1..8 stay in-group).
#pragma unroll
    for (int r = 0; r < 8; ++r) {
      float sa = pA[r], sb = pB[r];
#pragma unroll
      for (int sh = 1; sh < 16; sh <<= 1) {
        sa += __shfl_xor(sa, sh, 32);
        sb += __shfl_xor(sb, sh, 32);
      }
      pA[r] = sa; pB[r] = sb;
    }

    const float PSCALE = 0.0022097086912079612f;  // 1/(sqrt(128)*40)
    const float SQ3INV = 0.57735026918962576f;
    if ((lane & 15) == 0) {                       // lanes 0 and 16 write
#pragma unroll
      for (int r = 0; r < 8; ++r) {
        int e = base + r + 8 * hi;
        if (e >= tileStart) {                     // exactly-once ownership
          float y0 = edge_attrs[(long)e * 4];
          float p  = (pA[r] * y0 + pB[r] * SQ3INV) * PSCALE;
          p_out[e] = p;
          atomicAdd(&multipoles[(long)rcv[r] * 4],  p);
          atomicAdd(&multipoles[(long)snd[r] * 4], -p);
        }
      }
    }
  }
}

// -------------------------------------------------------------------------
extern "C" void kernel_launch(void* const* d_in, const int* in_sizes, int n_in,
                              void* d_out, int out_size, void* d_ws, size_t ws_size,
                              hipStream_t stream) {
  (void)n_in; (void)out_size; (void)ws_size;
  const float* node_feats = (const float*)d_in[1];
  const float* edge_attrs = (const float*)d_in[2];
  const float* edge_feats = (const float*)d_in[3];
  const int*   edge_index = (const int*)  d_in[4];
  const float* W10 = (const float*)d_in[8];
  const float* W11 = (const float*)d_in[9];
  const float* W20 = (const float*)d_in[10];
  const float* W21 = (const float*)d_in[11];
  const float* fw0 = (const float*)d_in[12];
  const float* fw1 = (const float*)d_in[13];
  const float* fw2 = (const float*)d_in[14];
  const float* fw3 = (const float*)d_in[15];
  const float* Wf  = (const float*)d_in[16];
  const float* Wm1 = (const float*)d_in[17];

  const int N = in_sizes[1] / 256;   // node_feats is (N, 256)
  const int E = in_sizes[3] / 8;     // edge_feats is (E, 8)

  float* s1t = (float*)d_ws;
  float* s2t = s1t + (size_t)N * 64;
  float* v1t = s2t + (size_t)N * 64;
  float* v2t = v1t + (size_t)N * 192;

  float* multipoles = (float*)d_out;             // (N, 4)
  float* p_out = multipoles + (size_t)N * 4;     // (E,)

  const int nodeTiles = (N + 15) / 16;
  const int edgeTiles = (E + 15) / 16;

  hipLaunchKernelGGL(node_linear_kernel, dim3((nodeTiles + 3) / 4), dim3(128), 0, stream,
                     node_feats, W10, W11, W20, W21, s1t, s2t, v1t, v2t, N, nodeTiles);
  hipLaunchKernelGGL(mvec_init_kernel, dim3((N + 255) / 256), dim3(256), 0, stream,
                     node_feats, Wm1, multipoles, N);
  hipLaunchKernelGGL(edge_kernel, dim3((edgeTiles + 3) / 4), dim3(128), 0, stream,
                     edge_attrs, edge_feats, edge_index, fw0, fw1, fw2, fw3, Wf,
                     s1t, s2t, v1t, v2t, multipoles, p_out, N, E, edgeTiles);
}
